// EfficientLocalMultiheadAttention_57372173140157
// MI455X (gfx1250) — matrix operand order repaired and checked by reading, not compiled, not run on hardware
//
#include <hip/hip_runtime.h>

#define BB   2
#define LL   2048
#define DD   1024
#define HH   16
#define DHH  64
#define NN   (BB*LL)
#define WINR 64
#define KSZ  129
#define KPAD 72   // LDS row pitch (ushorts): 144B = 36 banks -> conflict-free b128

#if __has_builtin(__builtin_amdgcn_global_load_async_to_lds_b128) && \
    __has_builtin(__builtin_amdgcn_s_wait_asynccnt)
#define ELMA_ASYNC 1
#else
#define ELMA_ASYNC 0
#endif

typedef __attribute__((ext_vector_type(16))) __bf16          v16bf;
typedef __attribute__((ext_vector_type(8)))  float           v8f;
typedef __attribute__((ext_vector_type(4)))  float           v4f;
typedef __attribute__((ext_vector_type(16))) unsigned short  v16us;
typedef __attribute__((ext_vector_type(8)))  unsigned short  v8us;
typedef __attribute__((ext_vector_type(4)))  unsigned short  v4us;

// exact parameter types for the async-LDS builtin (from clang's diagnostic):
// arg0: addrspace(1) pointer to int4, arg1: addrspace(3) pointer to int4
typedef int v4i_vs __attribute__((vector_size(4 * sizeof(int))));
typedef __attribute__((address_space(1))) v4i_vs* async_gptr;
typedef __attribute__((address_space(3))) v4i_vs* async_lptr;

union BFU { v16us u; v16bf b; };
union BFS { __bf16 b; unsigned short u; };

__device__ __forceinline__ unsigned short f2bf(float f) {
  BFS cv; cv.b = (__bf16)f;          // hardware f32->bf16 convert
  return cv.u;
}
__device__ __forceinline__ float bf2f(unsigned short h) {
  return __uint_as_float(((unsigned int)h) << 16);
}

__device__ __forceinline__ v8f wmma_bf16(v16bf a, v16bf b, v8f c) {
  return __builtin_amdgcn_wmma_f32_16x16x32_bf16(
      /*neg_a=*/false, a, /*neg_b=*/false, b,
      /*c_mod=*/(short)0, c, /*reuse_a=*/false, /*reuse_b=*/false);
}

// A fragment (16-bit, 16x32): elems 0..7 = row[k0..k0+7], 8..15 = row[k0+16..k0+23]
// caller passes k0 = base + 8*(lane>=16)
__device__ __forceinline__ v16bf packA_bf16(const unsigned short* row, int k0) {
  v8us c0 = *(const v8us*)(row + k0);
  v8us c1 = *(const v8us*)(row + k0 + 16);
  BFU r;
#pragma unroll
  for (int i = 0; i < 8; ++i) { r.u[i] = c0[i]; r.u[8 + i] = c1[i]; }
  return r.b;
}

// B fragment (16-bit, 32x16): elems 0..7 = K[k0..k0+7], 8..15 = K[k0+16..k0+23], caller passes
// k0 = base + 8*(lane>=16). "row" is the memory row holding column n's K data.
__device__ __forceinline__ v16bf packB_bf16(const unsigned short* row, int k0) {
  v8us c0 = *(const v8us*)(row + k0);
  v8us c1 = *(const v8us*)(row + k0 + 16);
  BFU r;
#pragma unroll
  for (int i = 0; i < 8; ++i) { r.u[i] = c0[i]; r.u[8 + i] = c1[i]; }
  return r.b;
}

// 16-byte global -> LDS copy: async on CDNA5 (ASYNCcnt), sync fallback otherwise
__device__ __forceinline__ void cp16(const unsigned short* gsrc, unsigned short* ldst) {
#if ELMA_ASYNC
  __builtin_amdgcn_global_load_async_to_lds_b128(
      (async_gptr)(void*)gsrc, (async_lptr)(void*)ldst, 0, 0);
#else
  *(v8us*)ldst = *(const v8us*)gsrc;
#endif
}
__device__ __forceinline__ void cp_wait() {
#if ELMA_ASYNC
  __builtin_amdgcn_s_wait_asynccnt(0);
#endif
}

// ---------------------------------------------------------------------------
// Bulk f32 -> bf16 conversion (one pass; keeps conversion out of GEMM loops)
// ---------------------------------------------------------------------------
__global__ __launch_bounds__(256) void elma_cvt(
    const float* __restrict__ in, unsigned short* __restrict__ out, int nvec4) {
  const int i = blockIdx.x * 256 + threadIdx.x;
  if (i < nvec4) {
    v4f f = *(const v4f*)(in + 4 * (size_t)i);
    v4us u;
#pragma unroll
    for (int j = 0; j < 4; ++j) u[j] = f2bf(f[j]);
    *(v4us*)(out + 4 * (size_t)i) = u;
  }
}

// ---------------------------------------------------------------------------
// Projection GEMM: out = X @ W^T + b  (X bf16 [N,D], W bf16 [D,D]) -> bf16
// vt_mode==0: out[((b*H+h)*L + l)*Dh + dd]          (Q/K layout)
// vt_mode==1: out[((b*H+h)*Dh + dd)*L + l] * mask   (masked V-transposed)
// one wave computes a 16x64 tile; 8 waves / block; 512 blocks.
// ---------------------------------------------------------------------------
__global__ __launch_bounds__(256) void elma_proj(
    const unsigned short* __restrict__ X, const unsigned short* __restrict__ W,
    const float* __restrict__ bias, const unsigned char* __restrict__ mask,
    unsigned short* __restrict__ out, int vt_mode) {
  const int wi = threadIdx.x >> 5, lane = threadIdx.x & 31;
  const int hl = lane >> 4, ln = lane & 15;
  const int wid = blockIdx.x * 8 + wi;          // 0..4095
  const int mtile = wid >> 4;                   // 0..255  (16 rows each)
  const int n4 = wid & 15;                      // 0..15   (64 cols each)
  const int mrow = mtile * 16 + ln;
  const unsigned short* arow = X + (size_t)mrow * DD;
  const unsigned short* brow = W + (size_t)(n4 * 64 + ln) * DD;
  const int kA = hl * 8, kB = hl * 8;

  v8f acc[4] = {v8f{}, v8f{}, v8f{}, v8f{}};
  for (int kt = 0; kt < DD; kt += 32) {
    __builtin_prefetch(arow + kt + 64, 0, 1);
    // group the full load clause, then the 4 WMMAs
    v16bf a  = packA_bf16(arow, kt + kA);
    v16bf b0 = packB_bf16(brow + (size_t)0  * 16 * DD, kt + kB);
    v16bf b1 = packB_bf16(brow + (size_t)1  * 16 * DD, kt + kB);
    v16bf b2 = packB_bf16(brow + (size_t)2  * 16 * DD, kt + kB);
    v16bf b3 = packB_bf16(brow + (size_t)3  * 16 * DD, kt + kB);
    acc[0] = wmma_bf16(a, b0, acc[0]);
    acc[1] = wmma_bf16(a, b1, acc[1]);
    acc[2] = wmma_bf16(a, b2, acc[2]);
    acc[3] = wmma_bf16(a, b3, acc[3]);
  }
#pragma unroll
  for (int j = 0; j < 4; ++j) {
    const int col = n4 * 64 + j * 16 + ln;
    const float bv = bias[col];
    const int h = col >> 6, dd = col & 63;
#pragma unroll
    for (int r = 0; r < 8; ++r) {
      const int m = mtile * 16 + r + 8 * hl;
      const int b = m >> 11, l = m & (LL - 1);
      float v = acc[j][r] + bv;
      size_t idx;
      if (vt_mode) {
        v *= mask[b * LL + l] ? 1.0f : 0.0f;    // fold key mask into V
        idx = ((size_t)(b * HH + h) * DHH + dd) * LL + l;
      } else {
        idx = ((size_t)(b * HH + h) * LL + l) * DHH + dd;
      }
      out[idx] = f2bf(v);
    }
  }
}

// ---------------------------------------------------------------------------
// Flash attention: 8 waves/block share one (b,h); each wave owns a 16-row q
// tile and loops over all 2048 keys in chunks of 32. K/V chunks are staged in
// LDS once per block (async global->LDS, double-buffered) and consumed by all
// waves. gbuf = softmax(QK^T*s + mask) @ V, f32 [B,H,L,Dh].
// ---------------------------------------------------------------------------
__global__ __launch_bounds__(256) void elma_flash(
    const unsigned short* __restrict__ Qb, const unsigned short* __restrict__ Kb,
    const unsigned short* __restrict__ Vt, const unsigned char* __restrict__ mask,
    float* __restrict__ gbuf) {
  __shared__ unsigned short pP[8][16][32];        // per-wave P transpose staging
  __shared__ unsigned short kst[2][32][KPAD];     // K chunk: 32 keys x 64 d
  __shared__ unsigned short vst[2][DHH][KPAD];    // V chunk: 64 d x 32 keys
  const int tid = threadIdx.x;
  const int wi = tid >> 5, lane = tid & 31;
  const int hl = lane >> 4, ln = lane & 15;
  const int wq = blockIdx.x * 8 + wi;             // 0..4095
  const int bh = wq >> 7;                         // 0..31 (same for whole block)
  const int qbase = (wq & 127) * 16;
  const int b = bh >> 4;
  const int kA = hl * 8, kB = hl * 8;

  const unsigned short* qrow = Qb + ((size_t)bh * LL + qbase + ln) * DHH;
  const v16bf aq0 = packA_bf16(qrow, kA);         // d = 0..31
  const v16bf aq1 = packA_bf16(qrow, 32 + kA);    // d = 32..63
  const unsigned short* kbase_g = Kb + (size_t)bh * LL * DHH;
  const unsigned short* vbase_g = Vt + (size_t)bh * DHH * LL;

  // staging indices: 256 threads x 16B cover each 4KB tile
  const int krow_s = tid >> 3, kcol_s = (tid & 7) * 8;   // K: key row, d chunk
  const int vrow_s = tid >> 2, vcol_s = (tid & 3) * 8;   // V: d row, key chunk

  float m8[8], l8[8];
#pragma unroll
  for (int r = 0; r < 8; ++r) { m8[r] = -3.0e38f; l8[r] = 0.f; }
  v8f O[4] = {v8f{}, v8f{}, v8f{}, v8f{}};

  // prologue: stage chunk 0
  cp16(kbase_g + (size_t)krow_s * DHH + kcol_s, &kst[0][krow_s][kcol_s]);
  cp16(vbase_g + (size_t)vrow_s * LL + vcol_s,  &vst[0][vrow_s][vcol_s]);
  cp_wait();
  __syncthreads();

  for (int kb = 0; kb < LL; kb += 32) {
    const int p = (kb >> 5) & 1;
    if (kb + 32 < LL) {   // stage next chunk into the other buffer
      cp16(kbase_g + (size_t)(kb + 32 + krow_s) * DHH + kcol_s, &kst[p ^ 1][krow_s][kcol_s]);
      cp16(vbase_g + (size_t)vrow_s * LL + (kb + 32) + vcol_s,  &vst[p ^ 1][vrow_s][vcol_s]);
    }
    v8f S[2];
#pragma unroll
    for (int c = 0; c < 2; ++c) {
      const unsigned short* krow = &kst[p][c * 16 + ln][0];
      v8f s = {};
      s = wmma_bf16(aq0, packB_bf16(krow, kB), s);
      s = wmma_bf16(aq1, packB_bf16(krow, 32 + kB), s);
      const float mk = mask[b * LL + kb + c * 16 + ln] ? 0.f : -1.0e30f;
#pragma unroll
      for (int r = 0; r < 8; ++r) s[r] = s[r] * 0.125f + mk;  // *1/sqrt(Dh)
      S[c] = s;
    }
    // online softmax; a row lives in one 16-lane half -> xor 1,2,4,8 reduces
    v8f P0, P1, cv;
#pragma unroll
    for (int r = 0; r < 8; ++r) {
      float rm = fmaxf(S[0][r], S[1][r]);
      rm = fmaxf(rm, __shfl_xor(rm, 1, 32));
      rm = fmaxf(rm, __shfl_xor(rm, 2, 32));
      rm = fmaxf(rm, __shfl_xor(rm, 4, 32));
      rm = fmaxf(rm, __shfl_xor(rm, 8, 32));
      const float mn = fmaxf(m8[r], rm);
      const float c0 = __expf(m8[r] - mn);
      const float p0 = __expf(S[0][r] - mn);
      const float p1 = __expf(S[1][r] - mn);
      float rs = p0 + p1;
      rs += __shfl_xor(rs, 1, 32);
      rs += __shfl_xor(rs, 2, 32);
      rs += __shfl_xor(rs, 4, 32);
      rs += __shfl_xor(rs, 8, 32);
      l8[r] = l8[r] * c0 + rs;
      m8[r] = mn;
      cv[r] = c0;
      P0[r] = p0; P1[r] = p1;
    }
#pragma unroll
    for (int t = 0; t < 4; ++t) O[t] *= cv;
    // C-layout P -> A-layout via wave-private LDS (DS ops are in-order per wave)
#pragma unroll
    for (int r = 0; r < 8; ++r) {
      pP[wi][r + 8 * hl][ln]      = f2bf(P0[r]);
      pP[wi][r + 8 * hl][16 + ln] = f2bf(P1[r]);
    }
    const unsigned short* pbase = &pP[wi][0][0];
    const v16bf pA = packA_bf16(pbase + ln * 32, kA);
#pragma unroll
    for (int t = 0; t < 4; ++t) {
      O[t] = wmma_bf16(pA, packB_bf16(&vst[p][t * 16 + ln][0], kB), O[t]);
    }
    cp_wait();          // our next-chunk async copies have landed
    __syncthreads();    // everyone done reading buf p and writing buf p^1
  }
#pragma unroll
  for (int r = 0; r < 8; ++r) {
    const float inv = 1.0f / fmaxf(l8[r], 1e-20f);
    float* orow = gbuf + ((size_t)bh * LL + qbase + r + 8 * hl) * DHH;
#pragma unroll
    for (int t = 0; t < 4; ++t) orow[t * 16 + ln] = O[t][r] * inv;
  }
}

// ---------------------------------------------------------------------------
// Local masked depthwise conv + mask-normalizer + per-head gate blend.
// heads bf16 [B, L, H*Dh]  =  alpha*global + (1-alpha)*local
// ---------------------------------------------------------------------------
__global__ __launch_bounds__(256) void elma_local(
    const unsigned short* __restrict__ Vt, const float* __restrict__ gbuf,
    const unsigned char* __restrict__ mask, const float* __restrict__ klog,
    const float* __restrict__ alog, unsigned short* __restrict__ heads) {
  __shared__ float kern[KSZ];
  const int bh = blockIdx.x, b = bh >> 4, h = bh & 15;
  const int tid = threadIdx.x;
  if (tid < KSZ) kern[tid] = 1.f / (1.f + __expf(-klog[h * KSZ + tid]));
  __syncthreads();
  const float alpha = 1.f / (1.f + __expf(-alog[h]));
  const unsigned short* vb = Vt + (size_t)bh * DHH * LL;

  for (int l = tid; l < LL; l += 256) {
    const int jlo = (l < WINR) ? (WINR - l) : 0;
    const int jhi = (l > LL - 1 - WINR) ? (LL - 1 - l + WINR) : (KSZ - 1);
    float den = 0.f;
    for (int j = jlo; j <= jhi; ++j)
      den += kern[j] * (mask[b * LL + l - WINR + j] ? 1.f : 0.f);
    const float invden = 1.f / fmaxf(den, 1e-6f);
    const float* grow = gbuf + ((size_t)bh * LL + l) * DHH;
    unsigned short* hrow = heads + ((size_t)(b * LL + l)) * DD + h * DHH;
    for (int d = 0; d < DHH; ++d) {
      const unsigned short* vrow = vb + (size_t)d * LL + (l - WINR);
      float acc = 0.f;
      for (int j = jlo; j <= jhi; ++j) acc += kern[j] * bf2f(vrow[j]);
      const float val = alpha * grow[d] + (1.f - alpha) * (acc * invden);
      hrow[d] = f2bf(val);
    }
  }
}

// ---------------------------------------------------------------------------
// Output GEMM: out = heads(bf16) @ Wo^T(bf16) + bo  -> f32 d_out [B,L,D]
// ---------------------------------------------------------------------------
__global__ __launch_bounds__(256) void elma_out(
    const unsigned short* __restrict__ A, const unsigned short* __restrict__ W,
    const float* __restrict__ bias, float* __restrict__ out) {
  const int wi = threadIdx.x >> 5, lane = threadIdx.x & 31;
  const int hl = lane >> 4, ln = lane & 15;
  const int wid = blockIdx.x * 8 + wi;
  const int mtile = wid >> 4, n4 = wid & 15;
  const int mrow = mtile * 16 + ln;
  const unsigned short* arow = A + (size_t)mrow * DD;
  const unsigned short* brow = W + (size_t)(n4 * 64 + ln) * DD;
  const int kA = hl * 8, kB = hl * 8;

  v8f acc[4] = {v8f{}, v8f{}, v8f{}, v8f{}};
  for (int kt = 0; kt < DD; kt += 32) {
    __builtin_prefetch(arow + kt + 64, 0, 1);
    v16bf a  = packA_bf16(arow, kt + kA);
    v16bf b0 = packB_bf16(brow + (size_t)0  * 16 * DD, kt + kB);
    v16bf b1 = packB_bf16(brow + (size_t)1  * 16 * DD, kt + kB);
    v16bf b2 = packB_bf16(brow + (size_t)2  * 16 * DD, kt + kB);
    v16bf b3 = packB_bf16(brow + (size_t)3  * 16 * DD, kt + kB);
    acc[0] = wmma_bf16(a, b0, acc[0]);
    acc[1] = wmma_bf16(a, b1, acc[1]);
    acc[2] = wmma_bf16(a, b2, acc[2]);
    acc[3] = wmma_bf16(a, b3, acc[3]);
  }
#pragma unroll
  for (int j = 0; j < 4; ++j) {
    const int col = n4 * 64 + j * 16 + ln;
    const float bv = bias[col];
#pragma unroll
    for (int r = 0; r < 8; ++r) {
      const int m = mtile * 16 + r + 8 * hl;
      out[(size_t)m * DD + col] = acc[j][r] + bv;
    }
  }
}

extern "C" void kernel_launch(void* const* d_in, const int* in_sizes, int n_in,
                              void* d_out, int out_size, void* d_ws, size_t ws_size,
                              hipStream_t stream) {
  const float* queries        = (const float*)d_in[0];
  const unsigned char* maskp  = (const unsigned char*)d_in[1];  // bool bytes
  const float* Wq = (const float*)d_in[2];
  const float* bq = (const float*)d_in[3];
  const float* Wk = (const float*)d_in[4];
  const float* bk = (const float*)d_in[5];
  const float* Wv = (const float*)d_in[6];
  const float* bv = (const float*)d_in[7];
  const float* Wo = (const float*)d_in[8];
  const float* bo = (const float*)d_in[9];
  const float* klog = (const float*)d_in[10];
  const float* alog = (const float*)d_in[11];

  char* ws = (char*)d_ws;
  unsigned short* Qb    = (unsigned short*)(ws);                       // 8 MB
  unsigned short* Kb    = (unsigned short*)(ws + ((size_t)8  << 20));  // 8 MB
  unsigned short* Vt    = (unsigned short*)(ws + ((size_t)16 << 20));  // 8 MB
  float*          gbuf  = (float*)        (ws + ((size_t)24 << 20));   // 16 MB
  unsigned short* heads = (unsigned short*)(ws + ((size_t)40 << 20));  // 8 MB
  unsigned short* Xb    = (unsigned short*)(ws + ((size_t)48 << 20));  // 8 MB
  unsigned short* Wqb   = (unsigned short*)(ws + ((size_t)56 << 20));  // 2 MB
  unsigned short* Wkb   = (unsigned short*)(ws + ((size_t)58 << 20));  // 2 MB
  unsigned short* Wvb   = (unsigned short*)(ws + ((size_t)60 << 20));  // 2 MB
  unsigned short* Wob   = (unsigned short*)(ws + ((size_t)62 << 20));  // 2 MB

  dim3 blk(256);
  // one-time f32 -> bf16 conversion passes (bandwidth-trivial, L2-resident)
  elma_cvt<<<(NN * DD / 4 + 255) / 256, blk, 0, stream>>>(queries, Xb, NN * DD / 4);
  elma_cvt<<<(DD * DD / 4 + 255) / 256, blk, 0, stream>>>(Wq, Wqb, DD * DD / 4);
  elma_cvt<<<(DD * DD / 4 + 255) / 256, blk, 0, stream>>>(Wk, Wkb, DD * DD / 4);
  elma_cvt<<<(DD * DD / 4 + 255) / 256, blk, 0, stream>>>(Wv, Wvb, DD * DD / 4);
  elma_cvt<<<(DD * DD / 4 + 255) / 256, blk, 0, stream>>>(Wo, Wob, DD * DD / 4);

  elma_proj <<<512, blk, 0, stream>>>(Xb, Wqb, bq, maskp, Qb, 0);
  elma_proj <<<512, blk, 0, stream>>>(Xb, Wkb, bk, maskp, Kb, 0);
  elma_proj <<<512, blk, 0, stream>>>(Xb, Wvb, bv, maskp, Vt, 1);
  elma_flash<<<512, blk, 0, stream>>>(Qb, Kb, Vt, maskp, gbuf);
  elma_local<<<32,  blk, 0, stream>>>(Vt, gbuf, maskp, klog, alog, heads);
  elma_out  <<<512, blk, 0, stream>>>(heads, Wob, bo, (float*)d_out);
}
